// GMemLinear_63565515981003
// MI455X (gfx1250) — matrix operand order repaired and checked by reading, not compiled, not run on hardware
//
#include <hip/hip_runtime.h>
#include <hip/hip_bf16.h>
#include <stdint.h>

// ---------------------------------------------------------------------------
// Problem constants (match reference)
// ---------------------------------------------------------------------------
#define TOKENS 8192
#define IN_F   4096
#define OUT_F  4096

// GEMM tiling
#define BM 128
#define BN 128
#define BK 32
#define LDSW 40                    // padded LDS row stride in bf16 (80 B = 64B row + 16B TDM pad)
#define ARR_E  (BM * LDSW)         // 5120 bf16 per array per buffer
#define ARR_B  (ARR_E * 2)         // 10240 bytes
#define BUF_E  (4 * ARR_E)         // Xh, Xl, Wh, Wl
#define BUF_B  (4 * ARR_B)         // 40960 bytes

typedef __attribute__((ext_vector_type(16))) __bf16 v16bf;
typedef __attribute__((ext_vector_type(8)))  __bf16 v8bf;
typedef __attribute__((ext_vector_type(8)))  float  v8f;
typedef __attribute__((ext_vector_type(4)))  unsigned int v4u;
typedef __attribute__((ext_vector_type(8)))  int v8i;
typedef __attribute__((ext_vector_type(4)))  int v4i;

// ---------------------------------------------------------------------------
// JAX threefry2x32 (exact): 5 groups of 4 rounds, key-schedule injection
// ---------------------------------------------------------------------------
__device__ __forceinline__ void threefry2x32(uint32_t k0, uint32_t k1,
                                             uint32_t x0, uint32_t x1,
                                             uint32_t& o0, uint32_t& o1) {
  uint32_t ks[3] = {k0, k1, k0 ^ k1 ^ 0x1BD11BDAu};
  x0 += ks[0]; x1 += ks[1];
  const int rot[2][4] = {{13, 15, 26, 6}, {17, 29, 16, 24}};
#pragma unroll
  for (int i = 0; i < 5; ++i) {
#pragma unroll
    for (int j = 0; j < 4; ++j) {
      const int r = rot[i & 1][j];
      x0 += x1;
      x1 = (x1 << r) | (x1 >> (32 - r));
      x1 ^= x0;
    }
    x0 += ks[(i + 1) % 3];
    x1 += ks[(i + 2) % 3] + (uint32_t)(i + 1);
  }
  o0 = x0; o1 = x1;
}

// bits -> uniform(-bound, bound), bound = 1/sqrt(4096) = 0.015625
__device__ __forceinline__ float bits_to_uniform(uint32_t b) {
  float u = __uint_as_float((b >> 9) | 0x3F800000u) - 1.0f;   // [0,1)
  float v = fmaf(u, 0.03125f, -0.015625f);                    // [-b, b)
  return fmaxf(v, -0.015625f);
}

__device__ __forceinline__ void split1(float v, __bf16& h, __bf16& l) {
  h = (__bf16)v;
  l = (__bf16)(v - (float)h);
}

// ---------------------------------------------------------------------------
// Kernel 1: hydrate virtual weight, emitting bf16 hi/lo split directly.
// JAX counter layout: element j pairs with j + N/2; out0 -> j, out1 -> j+N/2.
// ---------------------------------------------------------------------------
__global__ __launch_bounds__(256) void hydrate_w_split(
    __bf16* __restrict__ Wh, __bf16* __restrict__ Wl,
    const int* __restrict__ seed_p) {
  const uint32_t half = (uint32_t)(((uint64_t)OUT_F * IN_F) / 2);  // 8388608
  uint32_t j = blockIdx.x * blockDim.x + threadIdx.x;
  if (j >= half) return;
  uint32_t k1 = (uint32_t)seed_p[0];   // jax.random.key(seed): (hi=0, lo=seed)
  uint32_t o0, o1;
  threefry2x32(0u, k1, j, j + half, o0, o1);
  float v0 = bits_to_uniform(o0), v1 = bits_to_uniform(o1);
  __bf16 h, l;
  split1(v0, h, l); Wh[j] = h;        Wl[j] = l;
  split1(v1, h, l); Wh[j + half] = h; Wl[j + half] = l;
}

// ---------------------------------------------------------------------------
// Kernel 2: split x (f32) into bf16 hi + residual lo arrays (one pass).
// ---------------------------------------------------------------------------
__global__ __launch_bounds__(256) void split_x(const float* __restrict__ X,
                                               __bf16* __restrict__ Xh,
                                               __bf16* __restrict__ Xl) {
  size_t i0 = ((size_t)blockIdx.x * blockDim.x + threadIdx.x) * 8;
  float4 a = ((const float4*)(X + i0))[0];
  float4 b = ((const float4*)(X + i0))[1];
  float v[8] = {a.x, a.y, a.z, a.w, b.x, b.y, b.z, b.w};
  v8bf h, l;
#pragma unroll
  for (int j = 0; j < 8; ++j) { __bf16 hb, lb; split1(v[j], hb, lb); h[j] = hb; l[j] = lb; }
  *(v8bf*)(Xh + i0) = h;
  *(v8bf*)(Xl + i0) = l;
}

// ---------------------------------------------------------------------------
// Tensor Data Mover: one descriptor moves a 128x32-bf16 tile into LDS with
// hardware padding (row = 32 bf16 = 16 DWORDs, +4 DWORDs pad -> 80 B stride).
// D# layout per CDNA5 ISA ch.8.3/8.4 (group2/3 unused: tile_dim2..4 = 0).
// This toolchain's builtin is the 6-arg clang-23 form:
//   (uint32x4 g0, int32x8 g1, int32x4, int32x4, int32x8, i32 cpol)
// ---------------------------------------------------------------------------
__device__ __forceinline__ void tdm_load_tile(uint32_t lds_addr,
                                              const void* gaddr,
                                              uint32_t tensor_rows) {
  uint64_t ga = (uint64_t)(uintptr_t)gaddr;
  v4u g0;
  g0[0] = 1u;                                        // count=1, user mode, no gather
  g0[1] = lds_addr;                                  // LDS byte address
  g0[2] = (uint32_t)ga;                              // global_addr[31:0]
  g0[3] = (uint32_t)(ga >> 32) | (2u << 30);         // global_addr[56:32] | type=2
  v8i g1;
  g1[0] = (1 << 16)                                  // data_size = 2 bytes
        | (1 << 20)                                  // pad_enable
        | (3 << 22)                                  // pad_interval: every 16 DWORDs
        | (3 << 25);                                 // pad_amount: 4 DWORDs
  g1[1] = (int)((IN_F & 0xffffu) << 16);             // tensor_dim0 lo16 (=4096)
  g1[2] = (int)((tensor_rows & 0xffffu) << 16);      // dim0 hi16=0 | tensor_dim1 lo16
  g1[3] = (int)((32u << 16) | (tensor_rows >> 16));  // tensor_dim1 hi16 | tile_dim0=32
  g1[4] = 128;                                       // tile_dim1=128 | tile_dim2=0
  g1[5] = IN_F;                                      // tensor_dim0_stride lo32
  g1[6] = 0;                                         // dim0_stride hi16 | dim1_stride lo16
  g1[7] = 0;
  v4i z4 = {0, 0, 0, 0};
  v8i z8 = {0, 0, 0, 0, 0, 0, 0, 0};
  __builtin_amdgcn_tensor_load_to_lds(g0, g1, z4, z4, z8, 0);
}

// Wave 0 issues one K-chunk: 4 tiles (Xh, Xl, Wh, Wl) -> TENSORcnt += 4.
__device__ __forceinline__ void stage_tiles_tdm(
    uint32_t buf_off, const __bf16* __restrict__ Xh, const __bf16* __restrict__ Xl,
    const __bf16* __restrict__ Wh, const __bf16* __restrict__ Wl,
    int bm, int bn, int k0) {
  const size_t gx = (size_t)bm * IN_F + k0;
  const size_t gw = (size_t)bn * IN_F + k0;
  tdm_load_tile(buf_off + 0 * ARR_B, Xh + gx, TOKENS);
  tdm_load_tile(buf_off + 1 * ARR_B, Xl + gx, TOKENS);
  tdm_load_tile(buf_off + 2 * ARR_B, Wh + gw, OUT_F);
  tdm_load_tile(buf_off + 3 * ARR_B, Wl + gw, OUT_F);
}

// ---------------------------------------------------------------------------
// Kernel 3: Y[t][o] = sum_i x[t][i] * W[o][i] + bias[o]
// bf16x3 split GEMM on v_wmma_f32_16x16x32_bf16, double-buffered TDM staging.
//   block = 128x128 output tile, 256 threads = 8 waves (4 M x 2 N)
//   wave  = 32x64 (2 x 4 grid of 16x16 accumulators), 24 WMMA / K-chunk
// ---------------------------------------------------------------------------
__global__ __launch_bounds__(256) void gemm_bf16x3(
    const __bf16* __restrict__ Xh, const __bf16* __restrict__ Xl,
    const __bf16* __restrict__ Wh, const __bf16* __restrict__ Wl,
    const float* __restrict__ bias, float* __restrict__ Y) {
  __shared__ __bf16 smem[2 * BUF_E];                // 80 KB: ping-pong buffers
  const uint32_t lds_base = (uint32_t)(uintptr_t)(void*)&smem[0];

  const int bm = blockIdx.x * BM;
  const int bn = blockIdx.y * BN;
  const int tid   = threadIdx.x;
  const int lane  = tid & 31;
  const int wave  = tid >> 5;
  const int wm    = wave & 3;               // M sub-block: wm*32
  const int wn    = wave >> 2;              // N sub-block: wn*64
  const int lhalf = lane & 15;
  const int lhi   = lane >> 4;

  // Prologue: wave 0's TDM fills buffer 0 with the first K-chunk.
  if (wave == 0)
    stage_tiles_tdm(lds_base, Xh, Xl, Wh, Wl, bm, bn, 0);

  // Accumulators initialized with bias (overlaps with the DMA fill).
  v8f acc[2][4];
#pragma unroll
  for (int nt = 0; nt < 4; ++nt) {
    float b = bias[bn + wn * 64 + nt * 16 + lhalf];
#pragma unroll
    for (int mt = 0; mt < 2; ++mt)
#pragma unroll
      for (int r = 0; r < 8; ++r) acc[mt][nt][r] = b;
  }

  int parity = 0;
  for (int k0 = 0; k0 < IN_F; k0 += BK) {
    // Wave 0: refill the other buffer (its readers finished last iteration;
    // trailing barrier below published that), then wait for the current
    // buffer's 4 TDM ops (TDM completes in order per wave), then barrier to
    // publish the DMA-written LDS to all waves.
    if (wave == 0) {
      if (k0 + BK < IN_F) {
        stage_tiles_tdm(lds_base + (parity ^ 1) * BUF_B, Xh, Xl, Wh, Wl,
                        bm, bn, k0 + BK);
        __builtin_amdgcn_s_wait_tensorcnt(4);   // 4 newest = next buffer
      } else {
        __builtin_amdgcn_s_wait_tensorcnt(0);
      }
    }
    __syncthreads();                      // current-buffer tiles visible to all

    const __bf16* sb  = smem + parity * BUF_E;
    const __bf16* sXh = sb;
    const __bf16* sXl = sb + ARR_E;
    const __bf16* sWh = sb + 2 * ARR_E;
    const __bf16* sWl = sb + 3 * ARR_E;

    // A fragments (16-bit A 16x32 layout):
    // lane<16: M=lane, K={0..7,16..23}; lane>=16: M=lane-16, K={8..15,24..31}
    v16bf ah[2], al[2];
#pragma unroll
    for (int mt = 0; mt < 2; ++mt) {
      int arow = (wm * 32 + mt * 16 + lhalf) * LDSW;
      v8bf h0 = *(const v8bf*)&sXh[arow + lhi * 8];
      v8bf h1 = *(const v8bf*)&sXh[arow + 16 + lhi * 8];
      v8bf l0 = *(const v8bf*)&sXl[arow + lhi * 8];
      v8bf l1 = *(const v8bf*)&sXl[arow + 16 + lhi * 8];
      ah[mt] = __builtin_shufflevector(h0, h1, 0,1,2,3,4,5,6,7,8,9,10,11,12,13,14,15);
      al[mt] = __builtin_shufflevector(l0, l1, 0,1,2,3,4,5,6,7,8,9,10,11,12,13,14,15);
    }

    // B fragments (16-bit B 32x16 layout): lane<16: K={0..7,16..23}; lane>=16: K={8..15,24..31}
#pragma unroll
    for (int nt = 0; nt < 4; ++nt) {
      int brow = (wn * 64 + nt * 16 + lhalf) * LDSW;
      v8bf bh0 = *(const v8bf*)&sWh[brow + lhi * 8];
      v8bf bh1 = *(const v8bf*)&sWh[brow + 16 + lhi * 8];
      v8bf bl0 = *(const v8bf*)&sWl[brow + lhi * 8];
      v8bf bl1 = *(const v8bf*)&sWl[brow + 16 + lhi * 8];
      v16bf bh = __builtin_shufflevector(bh0, bh1, 0,1,2,3,4,5,6,7,8,9,10,11,12,13,14,15);
      v16bf bl = __builtin_shufflevector(bl0, bl1, 0,1,2,3,4,5,6,7,8,9,10,11,12,13,14,15);

      // x*w ~= xh*wh + xh*wl + xl*wh   (xl*wl ~ 2^-32, dropped)
#pragma unroll
      for (int mt = 0; mt < 2; ++mt)
        acc[mt][nt] = __builtin_amdgcn_wmma_f32_16x16x32_bf16(
            false, ah[mt], false, bh, (short)0, acc[mt][nt], false, false);
#pragma unroll
      for (int mt = 0; mt < 2; ++mt)
        acc[mt][nt] = __builtin_amdgcn_wmma_f32_16x16x32_bf16(
            false, ah[mt], false, bl, (short)0, acc[mt][nt], false, false);
#pragma unroll
      for (int mt = 0; mt < 2; ++mt)
        acc[mt][nt] = __builtin_amdgcn_wmma_f32_16x16x32_bf16(
            false, al[mt], false, bh, (short)0, acc[mt][nt], false, false);
    }
    __syncthreads();                 // all waves done reading current buffer
    parity ^= 1;
  }

  // Epilogue: C/D layout: lane<16 -> N=lane, M=r ; lane>=16 -> N=lane-16, M=r+8
#pragma unroll
  for (int mt = 0; mt < 2; ++mt) {
#pragma unroll
    for (int nt = 0; nt < 4; ++nt) {
      int col = bn + wn * 64 + nt * 16 + lhalf;
#pragma unroll
      for (int r = 0; r < 8; ++r) {
        int row = bm + wm * 32 + mt * 16 + lhi * 8 + r;
        Y[(size_t)row * OUT_F + col] = acc[mt][nt][r];
      }
    }
  }
}

// ---------------------------------------------------------------------------
// Launch.  Workspace layout (192 MiB):
//   Wh [OUT_F*IN_F] bf16 | Wl [OUT_F*IN_F] bf16 | Xh [TOKENS*IN_F] | Xl [...]
// ---------------------------------------------------------------------------
extern "C" void kernel_launch(void* const* d_in, const int* in_sizes, int n_in,
                              void* d_out, int out_size, void* d_ws, size_t ws_size,
                              hipStream_t stream) {
  const float* x    = (const float*)d_in[0];
  const float* bias = (const float*)d_in[1];
  const int*   seed = (const int*)d_in[2];
  float* Y = (float*)d_out;

  __bf16* Wh = (__bf16*)d_ws;
  __bf16* Wl = Wh + (size_t)OUT_F * IN_F;
  __bf16* Xh = Wl + (size_t)OUT_F * IN_F;
  __bf16* Xl = Xh + (size_t)TOKENS * IN_F;

  const uint32_t half = (uint32_t)(((uint64_t)OUT_F * IN_F) / 2);
  hydrate_w_split<<<(half + 255) / 256, 256, 0, stream>>>(Wh, Wl, seed);

  const size_t xelems = (size_t)TOKENS * IN_F;
  split_x<<<(unsigned)(xelems / 8 / 256), 256, 0, stream>>>(x, Xh, Xl);

  dim3 grid(TOKENS / BM, OUT_F / BN);      // 64 x 32
  gemm_bf16x3<<<grid, 256, 0, stream>>>(Xh, Xl, Wh, Wl, bias, Y);
}
